// GCN_Net_76819785056584
// MI455X (gfx1250) — compile-verified, matrix operand order checked by reading, not run on hardware
//
#include <hip/hip_runtime.h>
#include <math.h>

typedef __attribute__((ext_vector_type(16))) _Float16 v16h;
typedef __attribute__((ext_vector_type(8)))  float    v8f;

#define TAUF 0.05f
#define SKI 20
#define NB 512   // 16 wave32s, one WGP

// ~51 KB static LDS, reused across phases
struct Smem {
  float hid[64*64];       // node-MLP hidden; later reused as dinv[4096]
  float fpack[64*32*2];   // f1 | f2 (f32); later reused as the 64x64 sinkhorn matrix
  _Float16 ah[64*32];     // f1 normalized, f16 (WMMA A operand)
  _Float16 bh[64*32];     // f2 normalized, f16 (WMMA B operand)
  float cls1[64], cls2[64], cf1[64], cf2[64];
  float rowm[64], colm[64];
  float R[64*16];         // row partial sums of mu
  float Cc[64*16];        // col partial sums of mu
  float T[16];            // total sum of mu
  float Ssum;             // total class-match count
};

__device__ __forceinline__ void sinkhorn_block(float* M, int tid) {
  // M holds log_s (already divided by TAU). 20 alternating row/col LSE passes, then exp.
  for (int it = 0; it < SKI; ++it) {
    if (tid < 64) {
      if ((it & 1) == 0) {          // axis=1: normalize each row
        float mx = -1e30f;
        for (int j = 0; j < 64; ++j) mx = fmaxf(mx, M[tid*64+j]);
        float s = 0.f;
        for (int j = 0; j < 64; ++j) s += __expf(M[tid*64+j] - mx);
        float lse = mx + __logf(s);
        for (int j = 0; j < 64; ++j) M[tid*64+j] -= lse;
      } else {                      // axis=0: normalize each column
        float mx = -1e30f;
        for (int i = 0; i < 64; ++i) mx = fmaxf(mx, M[i*64+tid]);
        float s = 0.f;
        for (int i = 0; i < 64; ++i) s += __expf(M[i*64+tid] - mx);
        float lse = mx + __logf(s);
        for (int i = 0; i < 64; ++i) M[i*64+tid] -= lse;
      }
    }
    __syncthreads();
  }
  for (int idx = tid; idx < 4096; idx += NB) M[idx] = __expf(M[idx]);
  __syncthreads();
}

__global__ __launch_bounds__(NB)
void gcn_match_kernel(const float* __restrict__ ego, const float* __restrict__ cav,
                      const float* __restrict__ nw1, const float* __restrict__ nb1,
                      const float* __restrict__ nw2, const float* __restrict__ nb2,
                      const float* __restrict__ l0sw, const float* __restrict__ l0sb,
                      const float* __restrict__ l0cw, const float* __restrict__ l0cb,
                      const float* __restrict__ l0kw, const float* __restrict__ l0kb,
                      const float* __restrict__ l1sw, const float* __restrict__ l1sb,
                      const float* __restrict__ l1cw, const float* __restrict__ l1cb,
                      const float* __restrict__ l1kw, const float* __restrict__ l1kb,
                      const float* __restrict__ fcw, const float* __restrict__ fcb,
                      float* __restrict__ out, float* __restrict__ ws) {
  __shared__ Smem sm;
  const int tid  = (int)threadIdx.x;
  const int lane = tid & 31;
  const int wv   = tid >> 5;

  float* dinv  = sm.hid;                 // alias: valid after MLP phases
  float* f1    = sm.fpack;
  float* f2    = sm.fpack + 64*32;
  float* sinkM = sm.fpack;               // alias: valid after f16 conversion
  float* X  = ws;                        // 4096x16
  float* XN = ws + 65536;                // 4096x16
  float* Y  = ws + 131072;               // 4096x16
  float* x0 = ws + 196608;               // 4096 (diag(K) = node_aff)

  // hint the weights into cache (emits global_prefetch_b8)
  __builtin_prefetch(nw1, 0, 0);
  __builtin_prefetch(nw2, 0, 0);
  __builtin_prefetch(l1sw, 0, 0);
  __builtin_prefetch(l1cw, 0, 0);

  // ---- P0: classes and confidences --------------------------------------
  if (tid < 64) { sm.cls1[tid] = ego[tid*8+6]; sm.cf1[tid] = sqrtf(ego[tid*8+7]); }
  else if (tid < 128) { int t = tid-64; sm.cls2[t] = cav[t*8+6]; sm.cf2[t] = sqrtf(cav[t*8+7]); }
  __syncthreads();

  // ---- P1-P4: node MLP for ego and cav ----------------------------------
  for (int q = tid; q < 4096; q += NB) {           // hidden_ego = relu(x5 @ W1 + b1)
    int a = q >> 6, h = q & 63;
    float s = nb1[h];
    for (int k = 0; k < 5; ++k) s += ego[a*8+1+k] * nw1[k*64+h];
    sm.hid[q] = fmaxf(s, 0.f);
  }
  __syncthreads();
  for (int q = tid; q < 2048; q += NB) {           // f1 = hidden @ W2 + b2
    int a = q >> 5, o = q & 31;
    float s = nb2[o];
    for (int h = 0; h < 64; ++h) s += sm.hid[a*64+h] * nw2[h*32+o];
    f1[q] = s;
  }
  __syncthreads();
  for (int q = tid; q < 4096; q += NB) {           // hidden_cav
    int a = q >> 6, h = q & 63;
    float s = nb1[h];
    for (int k = 0; k < 5; ++k) s += cav[a*8+1+k] * nw1[k*64+h];
    sm.hid[q] = fmaxf(s, 0.f);
  }
  __syncthreads();
  for (int q = tid; q < 2048; q += NB) {           // f2
    int a = q >> 5, o = q & 31;
    float s = nb2[o];
    for (int h = 0; h < 64; ++h) s += sm.hid[a*64+h] * nw2[h*32+o];
    f2[q] = s;
  }
  __syncthreads();

  // ---- P5: row-normalize and convert to f16 for WMMA --------------------
  if (tid < 64) {
    float s = 0.f;
    for (int o = 0; o < 32; ++o) { float v = f1[tid*32+o]; s += v*v; }
    float inv = 1.f / fmaxf(sqrtf(s), 1e-8f);
    for (int o = 0; o < 32; ++o) sm.ah[tid*32+o] = (_Float16)(f1[tid*32+o] * inv);
  } else if (tid < 128) {
    int t = tid - 64;
    float s = 0.f;
    for (int o = 0; o < 32; ++o) { float v = f2[t*32+o]; s += v*v; }
    float inv = 1.f / fmaxf(sqrtf(s), 1e-8f);
    for (int o = 0; o < 32; ++o) sm.bh[t*32+o] = (_Float16)(f2[t*32+o] * inv);
  }
  __syncthreads();

  // ---- P6: cosine gram via WMMA; fuse node_aff = m * conf * cos ---------
  // 16 waves -> 16 tiles of 16x16 over K=32.
  {
    int mt = wv >> 2, nt = wv & 3;
    bool lo = lane < 16;
    int lr = lane & 15;
    int rA = mt*16 + lr;
    int rB = nt*16 + lr;
    int kb = lo ? 0 : 8;          // ISA 16-bit A layout K striping
    v16h a, b;
    for (int e = 0; e < 8; ++e) {
      a[e]   = sm.ah[rA*32 + kb + e];
      a[e+8] = sm.ah[rA*32 + kb + 16 + e];
      b[e]   = sm.bh[rB*32 + kb + e];
      b[e+8] = sm.bh[rB*32 + kb + 16 + e];
    }
    v8f c = {};
    c = __builtin_amdgcn_wmma_f32_16x16x32_f16(false, a, false, b, (short)0, c, false, false);
    for (int v = 0; v < 8; ++v) {
      int ai = mt*16 + v + (lo ? 0 : 8);   // C/D layout: VGPR v -> M=v (lanes<16) / v+8
      int bi = nt*16 + lr;
      float mf = (sm.cls1[ai] == sm.cls2[bi]) ? 1.f : 0.f;
      x0[ai*64+bi] = mf * sm.cf1[ai] * sm.cf2[bi] * c[v];
    }
  }

  // ---- P7: class-match mask marginals (structure of A) ------------------
  if (tid < 64) {
    float s = 0.f;
    for (int b2 = 0; b2 < 64; ++b2) s += (sm.cls1[tid]==sm.cls2[b2]) ? 1.f : 0.f;
    sm.rowm[tid] = s;
  } else if (tid < 128) {
    int t = tid - 64; float s = 0.f;
    for (int a2 = 0; a2 < 64; ++a2) s += (sm.cls1[a2]==sm.cls2[t]) ? 1.f : 0.f;
    sm.colm[t] = s;
  }
  __syncthreads();
  if (tid == 0) { float s = 0.f; for (int a2 = 0; a2 < 64; ++a2) s += sm.rowm[a2]; sm.Ssum = s; }
  __syncthreads();

  // ---- P8: dinv = rsqrt(A.sum(axis=1)) via rank structure ---------------
  for (int r = tid; r < 4096; r += NB) {
    int a = r >> 6, b2 = r & 63;
    float mf = (sm.cls1[a]==sm.cls2[b2]) ? 1.f : 0.f;
    float rowsum = 1.f + mf*(1.f + sm.Ssum - sm.rowm[a] - sm.colm[b2] + mf);
    dinv[r] = rsqrtf(rowsum);
  }
  __syncthreads();

  // ---- GCN layers -------------------------------------------------------
  for (int layer = 0; layer < 2; ++layer) {
    const float* sw = layer ? l1sw : l0sw;
    const float* sb = layer ? l1sb : l0sb;
    const float* cw = layer ? l1cw : l0cw;
    const float* cb = layer ? l1cb : l0cb;
    const float* kw = layer ? l1kw : l0kw;
    const float  kb0 = layer ? l1kb[0] : l0kb[0];

    if (layer == 0) {            // x is (4096,1): outer products
      for (int r = tid; r < 4096; r += NB) {
        float v = x0[r];
        for (int j = 0; j < 16; ++j) {
          XN[r*16+j] = v * sw[j];
          Y[r*16+j]  = v * cw[j];
        }
      }
    } else {                     // x(4096x16) @ sw / cw via WMMA, K=16 padded to 32
      bool lo = lane < 16;
      int ln = lane & 15;
      v16h bs, bc;
      for (int e = 0; e < 8; ++e) {
        int k = (lo ? 0 : 8) + e;
        bs[e]   = (_Float16)sw[k*16 + ln];
        bc[e]   = (_Float16)cw[k*16 + ln];
        bs[e+8] = (_Float16)0.f;
        bc[e+8] = (_Float16)0.f;
      }
      for (int t = wv; t < 256; t += 16) {
        int r0 = t*16;
        int rr = r0 + ln;
        v16h av;
        for (int e = 0; e < 8; ++e) {
          int k = (lo ? 0 : 8) + e;
          av[e]   = (_Float16)X[rr*16 + k];
          av[e+8] = (_Float16)0.f;
        }
        v8f cs = {}, cc = {};
        cs = __builtin_amdgcn_wmma_f32_16x16x32_f16(false, av, false, bs, (short)0, cs, false, false);
        cc = __builtin_amdgcn_wmma_f32_16x16x32_f16(false, av, false, bc, (short)0, cc, false, false);
        for (int v = 0; v < 8; ++v) {
          int r = r0 + v + (lo ? 0 : 8);
          XN[r*16 + ln] = cs[v];
          Y[r*16 + ln]  = cc[v];
        }
      }
    }
    __syncthreads();

    // separable partial sums of mu = m * dinv * (x@cw)  (replaces normA @ y)
    for (int q = tid; q < 1024; q += NB) {
      int a = q >> 4, j = q & 15;
      float c1 = sm.cls1[a];
      float s = 0.f;
      for (int b2 = 0; b2 < 64; ++b2)
        if (c1 == sm.cls2[b2]) { int r = a*64+b2; s += dinv[r] * Y[r*16+j]; }
      sm.R[q] = s;
    }
    for (int q = tid; q < 1024; q += NB) {
      int b2 = q >> 4, j = q & 15;
      float c2 = sm.cls2[b2];
      float s = 0.f;
      for (int a = 0; a < 64; ++a)
        if (sm.cls1[a] == c2) { int r = a*64+b2; s += dinv[r] * Y[r*16+j]; }
      sm.Cc[q] = s;
    }
    __syncthreads();
    if (tid < 16) {
      float s = 0.f;
      for (int a = 0; a < 64; ++a) s += sm.R[a*16+tid];
      sm.T[tid] = s;
    }
    __syncthreads();

    // combine: xnew = x@sw + sb + dinv .* (A .* dinv .* y summed) + cb ; sink input
    for (int r = tid; r < 4096; r += NB) {
      int a = r >> 6, b2 = r & 63;
      float mf = (sm.cls1[a]==sm.cls2[b2]) ? 1.f : 0.f;
      float dv = dinv[r];
      float sc = 0.f;
      for (int j = 0; j < 16; ++j) {
        float u  = dv * Y[r*16+j];
        float mu = mf * u;
        float conv = dv * ((mf + 1.f)*u + mf*(sm.T[j] - sm.R[a*16+j] - sm.Cc[b2*16+j]) + mu);
        float xv = XN[r*16+j] + sb[j] + cb[j] + conv;
        XN[r*16+j] = xv;
        sc += xv * kw[j];
      }
      sinkM[r] = (sc + kb0) * (1.f/TAUF);
    }
    __syncthreads();

    sinkhorn_block(sinkM, tid);

    for (int r = tid; r < 4096; r += NB) {
      float add = sinkM[r];
      for (int j = 0; j < 16; ++j) X[r*16+j] = XN[r*16+j] + add;
    }
    __syncthreads();
  }

  // ---- final scores, transpose, sinkhorn, output ------------------------
  for (int r = tid; r < 4096; r += NB) {
    float sc = fcb[0];
    for (int j = 0; j < 16; ++j) sc += X[r*16+j] * fcw[j];
    int a = r >> 6, b2 = r & 63;
    sinkM[b2*64 + a] = sc * (1.f/TAUF);   // s = scores.reshape(64,64).T
  }
  __syncthreads();
  sinkhorn_block(sinkM, tid);
  for (int idx = tid; idx < 4096; idx += NB) out[idx] = sinkM[idx];
}

extern "C" void kernel_launch(void* const* d_in, const int* in_sizes, int n_in,
                              void* d_out, int out_size, void* d_ws, size_t ws_size,
                              hipStream_t stream) {
  (void)in_sizes; (void)n_in; (void)out_size; (void)ws_size;
  const float* ego  = (const float*)d_in[0];
  const float* cav  = (const float*)d_in[1];
  const float* nw1  = (const float*)d_in[2];
  const float* nb1  = (const float*)d_in[3];
  const float* nw2  = (const float*)d_in[4];
  const float* nb2  = (const float*)d_in[5];
  // d_in[6..9] = edge MLP weights: dead (edge_aff only used as a nonzero mask)
  const float* l0sw = (const float*)d_in[10];
  const float* l0sb = (const float*)d_in[11];
  const float* l0cw = (const float*)d_in[12];
  const float* l0cb = (const float*)d_in[13];
  const float* l0kw = (const float*)d_in[14];
  const float* l0kb = (const float*)d_in[15];
  const float* l1sw = (const float*)d_in[16];
  const float* l1sb = (const float*)d_in[17];
  const float* l1cw = (const float*)d_in[18];
  const float* l1cb = (const float*)d_in[19];
  const float* l1kw = (const float*)d_in[20];
  const float* l1kb = (const float*)d_in[21];
  const float* fcw  = (const float*)d_in[22];
  const float* fcb  = (const float*)d_in[23];

  gcn_match_kernel<<<dim3(1), dim3(NB), 0, stream>>>(
      ego, cav, nw1, nb1, nw2, nb2,
      l0sw, l0sb, l0cw, l0cb, l0kw, l0kb,
      l1sw, l1sb, l1cw, l1cb, l1kw, l1kb,
      fcw, fcb, (float*)d_out, (float*)d_ws);
}
